// AttentionBlock_69904887710664
// MI455X (gfx1250) — matrix operand order repaired and checked by reading, not compiled, not run on hardware
//
#include <hip/hip_runtime.h>
#include <hip/hip_bf16.h>
#include <math.h>

typedef __attribute__((ext_vector_type(16))) _Float16 v16h;
typedef __attribute__((ext_vector_type(8)))  _Float16 v8h;
typedef __attribute__((ext_vector_type(8)))  float    v8f;

// Problem constants (fixed by reference): b=2, C=256, H=W=64 -> n=4096,
// heads=4, hd=64, groups=32 (8 ch/group), eps=1e-5, scale = 1/8.
#define BATCH 2
#define CDIM  256
#define NDIM  4096
#define HEADS 4
#define HD    64

__device__ __forceinline__ v16h cat8(v8h lo, v8h hi) {
  return __builtin_shufflevector(lo, hi, 0,1,2,3,4,5,6,7,8,9,10,11,12,13,14,15);
}

// A fragment (16 rows x 32 K), tile row-major with K contiguous, stride ld halves.
// ISA layout: lane<16 -> row=lane, elems {K=sel*8..+7} then {K=16+sel*8..+7}.
__device__ __forceinline__ v16h load_a_frag(const _Float16* tile, int ld, int lane, int k0) {
  const int row = lane & 15;
  const int sel = lane >> 4;
  const _Float16* p = tile + row * ld + k0 + sel * 8;
  v8h lo = *(const v8h*)p;
  v8h hi = *(const v8h*)(p + 16);
  return cat8(lo, hi);
}

// B fragment (32 K x 16 cols) read from a TRANSPOSED tile (rows = N, K contiguous).
// ISA layout: lane holds col=lane&15, elems {K=sel*8..+7} then {K=16+sel*8..+7}.
__device__ __forceinline__ v16h load_bt_frag(const _Float16* tileT, int ld, int lane, int c0, int k0) {
  const int col = c0 + (lane & 15);
  const int sel = lane >> 4;
  const _Float16* p = tileT + col * ld + k0 + sel * 8;
  v8h lo = *(const v8h*)p;
  v8h hi = *(const v8h*)(p + 16);
  return cat8(lo, hi);
}

__device__ __forceinline__ v8f wmma_f16(v16h a, v16h b, v8f c) {
  return __builtin_amdgcn_wmma_f32_16x16x32_f16(false, a, false, b, (short)0, c, false, false);
}

// ---------------------------------------------------------------------------
// Kernel 1: convert weights f32 -> f16 (w_qkv 768x256, w_proj 256x256)
// ---------------------------------------------------------------------------
__global__ void cvt_weights_kernel(const float* __restrict__ wq, const float* __restrict__ wp,
                                   _Float16* __restrict__ wq16, _Float16* __restrict__ wp16) {
  int i = blockIdx.x * blockDim.x + threadIdx.x;
  if (i < 768 * 256) wq16[i] = (_Float16)wq[i];
  if (i < 256 * 256) wp16[i] = (_Float16)wp[i];
}

// ---------------------------------------------------------------------------
// Kernel 2: GroupNorm.  One block per (batch, group): 8 ch * 4096 px = 32768.
// Writes hn as (b, n, C) f16  (n-major so GEMM B^T staging is contiguous).
// ---------------------------------------------------------------------------
__global__ void __launch_bounds__(256) gn_kernel(const float* __restrict__ x,
                                                 const float* __restrict__ gs,
                                                 const float* __restrict__ gb,
                                                 _Float16* __restrict__ hn) {
  __shared__ float red[256];
  __shared__ float red2[256];
  const int t  = threadIdx.x;
  const int bi = blockIdx.x >> 5;
  const int g  = blockIdx.x & 31;
  const float* xp = x + ((size_t)bi * CDIM + g * 8) * NDIM;

  float s = 0.f, sq = 0.f;
  for (int i = t; i < 8 * NDIM; i += 256) {
    float v = xp[i];
    s += v; sq += v * v;
  }
  red[t] = s; red2[t] = sq;
  __syncthreads();
  for (int off = 128; off > 0; off >>= 1) {
    if (t < off) { red[t] += red[t + off]; red2[t] += red2[t + off]; }
    __syncthreads();
  }
  const float inv_cnt = 1.0f / (8.0f * NDIM);
  float mean = red[0] * inv_cnt;
  float var  = red2[0] * inv_cnt - mean * mean;
  float rstd = rsqrtf(var + 1e-5f);

  for (int i = t; i < 8 * NDIM; i += 256) {
    int cc = i >> 12;          // channel within group
    int n  = i & (NDIM - 1);
    int c  = g * 8 + cc;
    float v = (xp[i] - mean) * rstd * gs[c] + gb[c];
    hn[((size_t)bi * NDIM + n) * CDIM + c] = (_Float16)v;
  }
}

// ---------------------------------------------------------------------------
// Kernel 3: QKV GEMM.  qkv[o,n] = sum_c w_qkv[o,c] * hn[c,n]  (M=768,K=256,N=4096)
// Block tile 128x128, 8 waves (4M x 2N), wave tile 32x64 = 2x4 WMMA tiles.
// K-step 64, software-pipelined staging (next chunk's global loads overlap WMMA).
// Epilogue scatters: q,k -> (b,h,n,d) (q pre-scaled by 1/8) ; v -> (b, c, n).
// ---------------------------------------------------------------------------
__global__ void __launch_bounds__(256) qkv_gemm_kernel(const _Float16* __restrict__ hn,
                                                       const _Float16* __restrict__ w,
                                                       _Float16* __restrict__ q_ws,
                                                       _Float16* __restrict__ k_ws,
                                                       _Float16* __restrict__ v_ws) {
  __shared__ __align__(16) _Float16 at[128 * 72];   // A tile rows (M), 64 K contiguous, pad 72
  __shared__ __align__(16) _Float16 bt[128 * 72];   // B^T tile rows (N), 64 K contiguous

  const int t = threadIdx.x;
  const int lane = t & 31, wv = t >> 5;
  const int wm = wv >> 1, wn = wv & 1;
  const int m0 = blockIdx.x * 128;
  const int n0 = blockIdx.y * 128;
  const int b  = blockIdx.z;
  const _Float16* hnb = hn + (size_t)b * NDIM * CDIM;

  v8f acc[2][4];
  v8f zero = {};
#pragma unroll
  for (int i = 0; i < 2; ++i)
#pragma unroll
    for (int j = 0; j < 4; ++j) acc[i][j] = zero;

  const int srow = t >> 1;
  const int soff = (t & 1) * 32;
  const _Float16* ga = w + (size_t)(m0 + srow) * CDIM + soff;
  const _Float16* gbp = hnb + (size_t)(n0 + srow) * CDIM + soff;
  v8h* ad = (v8h*)(at + srow * 72 + soff);
  v8h* bd = (v8h*)(bt + srow * 72 + soff);

  { // prologue: stage K chunk 0
    const v8h* as = (const v8h*)ga;
    const v8h* bs = (const v8h*)gbp;
#pragma unroll
    for (int i = 0; i < 4; ++i) { ad[i] = as[i]; bd[i] = bs[i]; }
  }
  __syncthreads();

  for (int k0 = 0; k0 < CDIM; k0 += 64) {
    // issue next chunk's global loads before compute (latency overlaps WMMA)
    v8h an[4], bn[4];
    const bool nxt = (k0 + 64) < CDIM;
    if (nxt) {
      const v8h* as = (const v8h*)(ga + k0 + 64);
      const v8h* bs = (const v8h*)(gbp + k0 + 64);
#pragma unroll
      for (int i = 0; i < 4; ++i) { an[i] = as[i]; bn[i] = bs[i]; }
    }

#pragma unroll
    for (int kc = 0; kc < 2; ++kc) {
      v16h af[2], bf[4];
#pragma unroll
      for (int im = 0; im < 2; ++im)
        af[im] = load_a_frag(at + (wm * 32 + im * 16) * 72, 72, lane, kc * 32);
#pragma unroll
      for (int jn = 0; jn < 4; ++jn)
        bf[jn] = load_bt_frag(bt + (wn * 64) * 72, 72, lane, jn * 16, kc * 32);
#pragma unroll
      for (int im = 0; im < 2; ++im)
#pragma unroll
        for (int jn = 0; jn < 4; ++jn)
          acc[im][jn] = wmma_f16(af[im], bf[jn], acc[im][jn]);
    }
    __syncthreads();                 // all reads of at/bt done
    if (nxt) {
#pragma unroll
      for (int i = 0; i < 4; ++i) { ad[i] = an[i]; bd[i] = bn[i]; }
    }
    __syncthreads();                 // stores visible for next iteration
  }

  const int crow = lane >> 4, ccol = lane & 15;
#pragma unroll
  for (int im = 0; im < 2; ++im)
#pragma unroll
    for (int jn = 0; jn < 4; ++jn)
#pragma unroll
      for (int r = 0; r < 8; ++r) {
        int o = m0 + wm * 32 + im * 16 + r + 8 * crow;
        int n = n0 + wn * 64 + jn * 16 + ccol;
        float val = acc[im][jn][r];
        if (o < 256) {               // q: pre-scaled, (b,h,n,d)
          int h = o >> 6, d = o & 63;
          q_ws[(((size_t)b * HEADS + h) * NDIM + n) * HD + d] = (_Float16)(val * 0.125f);
        } else if (o < 512) {        // k: transposed too -> (b,h,n,d)
          int oo = o - 256, h = oo >> 6, d = oo & 63;
          k_ws[(((size_t)b * HEADS + h) * NDIM + n) * HD + d] = (_Float16)val;
        } else {                     // v: natural (b, c, n)
          v_ws[((size_t)b * CDIM + (o - 512)) * NDIM + n] = (_Float16)val;
        }
      }
}

// ---------------------------------------------------------------------------
// Kernel 4: flash attention.  Block = (b, h, 128 q-rows); 8 waves, each wave
// owns a 16-row slab.  Streams 64-wide K/V tiles with online softmax.
// K is (n,d) so both Q and K tiles stage as contiguous b128 copies; V is (d,n)
// so the V^T B-fragments read contiguous K runs.  Software-pipelined staging.
// ---------------------------------------------------------------------------
__global__ void __launch_bounds__(256) flash_attn_kernel(const _Float16* __restrict__ q_ws,
                                                         const _Float16* __restrict__ k_ws,
                                                         const _Float16* __restrict__ v_ws,
                                                         _Float16* __restrict__ out_ws) {
  __shared__ __align__(16) _Float16 qt[128 * 72];       // Q^T tile: (n, d)
  __shared__ __align__(16) _Float16 kt[64 * 72];        // K^T tile: (m, d)
  __shared__ __align__(16) _Float16 vt[64 * 72];        // V tile:   (d, m)
  __shared__ __align__(16) _Float16 pl[8 * 16 * 72];    // per-wave P: (16 n, 64 m)

  const int t = threadIdx.x;
  const int lane = t & 31, wv = t >> 5;
  const int crow = lane >> 4, ccol = lane & 15;
  const int n0 = blockIdx.x * 128;
  const int h  = blockIdx.y;
  const int b  = blockIdx.z;

  const _Float16* qp = q_ws + (((size_t)b * HEADS + h) * NDIM) * HD;
  const _Float16* kp = k_ws + (((size_t)b * HEADS + h) * NDIM) * HD;
  const _Float16* vp = v_ws + ((size_t)b * CDIM + h * HD) * NDIM;

  { // stage Q tile once (rows n0..n0+127, 64 halves each)
    int row = t >> 1, doff = (t & 1) * 32;
    const v8h* src = (const v8h*)(qp + (size_t)(n0 + row) * HD + doff);
    v8h* dst = (v8h*)(qt + row * 72 + doff);
#pragma unroll
    for (int i = 0; i < 4; ++i) dst[i] = src[i];
  }
  // staging geometry for K/V tiles: 64x64 halves, 16 halves per thread
  const int srow = t >> 2, soff = (t & 3) * 16;
  v8h* kd = (v8h*)(kt + srow * 72 + soff);
  v8h* vd = (v8h*)(vt + srow * 72 + soff);
  { // stage first K/V tile
    const v8h* ks = (const v8h*)(kp + (size_t)srow * HD + soff);
    const v8h* vs = (const v8h*)(vp + (size_t)srow * NDIM + soff);
    kd[0] = ks[0]; kd[1] = ks[1];
    vd[0] = vs[0]; vd[1] = vs[1];
  }

  v8f oacc[4];
  v8f zero = {};
#pragma unroll
  for (int j = 0; j < 4; ++j) oacc[j] = zero;
  float mrow[8], lrow[8];
#pragma unroll
  for (int r = 0; r < 8; ++r) { mrow[r] = -__builtin_inff(); lrow[r] = 0.f; }

  __syncthreads();

  _Float16* pb = pl + wv * 16 * 72;

  for (int m0 = 0; m0 < NDIM; m0 += 64) {
    // issue next-tile global loads first; latency overlaps the WMMA block
    v8h kn0, kn1, vn0, vn1;
    const bool nxt = (m0 + 64) < NDIM;
    if (nxt) {
      const v8h* ks = (const v8h*)(kp + (size_t)(m0 + 64 + srow) * HD + soff);
      const v8h* vs = (const v8h*)(vp + (size_t)srow * NDIM + (m0 + 64) + soff);
      kn0 = ks[0]; kn1 = ks[1];
      vn0 = vs[0]; vn1 = vs[1];
      if (m0 + 128 < NDIM) {   // gfx1250 global_prefetch_b8 one tile further out
        __builtin_prefetch(kp + (size_t)(m0 + 128 + srow) * HD + soff, 0, 1);
        __builtin_prefetch(vp + (size_t)srow * NDIM + (m0 + 128) + soff, 0, 1);
      }
    }

    // S = Q^T(16x64) x K(64x64) for this wave's 16 rows
    v16h aq0 = load_a_frag(qt + (wv * 16) * 72, 72, lane, 0);
    v16h aq1 = load_a_frag(qt + (wv * 16) * 72, 72, lane, 32);
    v8f s[4];
#pragma unroll
    for (int j = 0; j < 4; ++j) {
      v8f z = zero;
      z = wmma_f16(aq0, load_bt_frag(kt, 72, lane, j * 16, 0), z);
      z = wmma_f16(aq1, load_bt_frag(kt, 72, lane, j * 16, 32), z);
      s[j] = z;
    }

    // online softmax: rows are (r + 8*crow); cols live inside a 16-lane group
    float pm[8];
#pragma unroll
    for (int r = 0; r < 8; ++r) {
      float m_ = fmaxf(fmaxf(s[0][r], s[1][r]), fmaxf(s[2][r], s[3][r]));
#pragma unroll
      for (int msk = 1; msk < 16; msk <<= 1) m_ = fmaxf(m_, __shfl_xor(m_, msk, 32));
      pm[r] = m_;
    }
    float alpha[8];
#pragma unroll
    for (int r = 0; r < 8; ++r) {
      float mn = fmaxf(mrow[r], pm[r]);
      alpha[r] = __expf(mrow[r] - mn);
      mrow[r] = mn;
    }
    float ps[8];
#pragma unroll
    for (int r = 0; r < 8; ++r) ps[r] = 0.f;
#pragma unroll
    for (int j = 0; j < 4; ++j)
#pragma unroll
      for (int r = 0; r < 8; ++r) {
        float p = __expf(s[j][r] - mrow[r]);
        ps[r] += p;
        pb[(r + 8 * crow) * 72 + j * 16 + ccol] = (_Float16)p;
      }
#pragma unroll
    for (int r = 0; r < 8; ++r) {
#pragma unroll
      for (int msk = 1; msk < 16; msk <<= 1) ps[r] += __shfl_xor(ps[r], msk, 32);
      lrow[r] = lrow[r] * alpha[r] + ps[r];
    }
#pragma unroll
    for (int j = 0; j < 4; ++j)
#pragma unroll
      for (int r = 0; r < 8; ++r) oacc[j][r] *= alpha[r];

    // O += P(16x64) x V^T(64x64)
    v16h pa0 = load_a_frag(pb, 72, lane, 0);
    v16h pa1 = load_a_frag(pb, 72, lane, 32);
#pragma unroll
    for (int jd = 0; jd < 4; ++jd) {
      oacc[jd] = wmma_f16(pa0, load_bt_frag(vt, 72, lane, jd * 16, 0), oacc[jd]);
      oacc[jd] = wmma_f16(pa1, load_bt_frag(vt, 72, lane, jd * 16, 32), oacc[jd]);
    }
    __syncthreads();                 // all reads of kt/vt done
    if (nxt) {
      kd[0] = kn0; kd[1] = kn1;
      vd[0] = vn0; vd[1] = vn1;
    }
    __syncthreads();                 // stores visible for next iteration
  }

  float inv[8];
#pragma unroll
  for (int r = 0; r < 8; ++r) inv[r] = 1.0f / lrow[r];
#pragma unroll
  for (int jd = 0; jd < 4; ++jd)
#pragma unroll
    for (int r = 0; r < 8; ++r) {
      int n = n0 + wv * 16 + r + 8 * crow;
      int c = h * HD + jd * 16 + ccol;
      out_ws[((size_t)b * NDIM + n) * CDIM + c] = (_Float16)(oacc[jd][r] * inv[r]);
    }
}

// ---------------------------------------------------------------------------
// Kernel 5: proj GEMM + bias + residual.  M=256, K=256, N=4096.
// Same pipelined structure as the QKV GEMM.
// ---------------------------------------------------------------------------
__global__ void __launch_bounds__(256) proj_gemm_kernel(const _Float16* __restrict__ ows,
                                                        const _Float16* __restrict__ w,
                                                        const float* __restrict__ bproj,
                                                        const float* __restrict__ x,
                                                        float* __restrict__ out) {
  __shared__ __align__(16) _Float16 at[128 * 72];
  __shared__ __align__(16) _Float16 bt[128 * 72];

  const int t = threadIdx.x;
  const int lane = t & 31, wv = t >> 5;
  const int wm = wv >> 1, wn = wv & 1;
  const int m0 = blockIdx.x * 128;
  const int n0 = blockIdx.y * 128;
  const int b  = blockIdx.z;
  const _Float16* ob = ows + (size_t)b * NDIM * CDIM;

  v8f acc[2][4];
  v8f zero = {};
#pragma unroll
  for (int i = 0; i < 2; ++i)
#pragma unroll
    for (int j = 0; j < 4; ++j) acc[i][j] = zero;

  const int srow = t >> 1;
  const int soff = (t & 1) * 32;
  const _Float16* ga = w + (size_t)(m0 + srow) * CDIM + soff;
  const _Float16* gbp = ob + (size_t)(n0 + srow) * CDIM + soff;
  v8h* ad = (v8h*)(at + srow * 72 + soff);
  v8h* bd = (v8h*)(bt + srow * 72 + soff);

  {
    const v8h* as = (const v8h*)ga;
    const v8h* bs = (const v8h*)gbp;
#pragma unroll
    for (int i = 0; i < 4; ++i) { ad[i] = as[i]; bd[i] = bs[i]; }
  }
  __syncthreads();

  for (int k0 = 0; k0 < CDIM; k0 += 64) {
    v8h an[4], bn[4];
    const bool nxt = (k0 + 64) < CDIM;
    if (nxt) {
      const v8h* as = (const v8h*)(ga + k0 + 64);
      const v8h* bs = (const v8h*)(gbp + k0 + 64);
#pragma unroll
      for (int i = 0; i < 4; ++i) { an[i] = as[i]; bn[i] = bs[i]; }
    }

#pragma unroll
    for (int kc = 0; kc < 2; ++kc) {
      v16h af[2], bf[4];
#pragma unroll
      for (int im = 0; im < 2; ++im)
        af[im] = load_a_frag(at + (wm * 32 + im * 16) * 72, 72, lane, kc * 32);
#pragma unroll
      for (int jn = 0; jn < 4; ++jn)
        bf[jn] = load_bt_frag(bt + (wn * 64) * 72, 72, lane, jn * 16, kc * 32);
#pragma unroll
      for (int im = 0; im < 2; ++im)
#pragma unroll
        for (int jn = 0; jn < 4; ++jn)
          acc[im][jn] = wmma_f16(af[im], bf[jn], acc[im][jn]);
    }
    __syncthreads();
    if (nxt) {
#pragma unroll
      for (int i = 0; i < 4; ++i) { ad[i] = an[i]; bd[i] = bn[i]; }
    }
    __syncthreads();
  }

  const int crow = lane >> 4, ccol = lane & 15;
#pragma unroll
  for (int im = 0; im < 2; ++im)
#pragma unroll
    for (int jn = 0; jn < 4; ++jn)
#pragma unroll
      for (int r = 0; r < 8; ++r) {
        int o = m0 + wm * 32 + im * 16 + r + 8 * crow;
        int n = n0 + wn * 64 + jn * 16 + ccol;
        size_t idx = ((size_t)b * CDIM + o) * NDIM + n;
        out[idx] = acc[im][jn][r] + bproj[o] + x[idx];
      }
}

// ---------------------------------------------------------------------------
// Launch
// ---------------------------------------------------------------------------
extern "C" void kernel_launch(void* const* d_in, const int* in_sizes, int n_in,
                              void* d_out, int out_size, void* d_ws, size_t ws_size,
                              hipStream_t stream) {
  const float* x     = (const float*)d_in[0];
  const float* gs    = (const float*)d_in[1];
  const float* gb    = (const float*)d_in[2];
  const float* wqkv  = (const float*)d_in[3];
  const float* wproj = (const float*)d_in[4];
  const float* bproj = (const float*)d_in[5];
  float* out = (float*)d_out;

  char* ws = (char*)d_ws;
  // f16 scratch regions (all 256B aligned):
  _Float16* hn   = (_Float16*)(ws);                        // (b, n, C)      4 MB
  _Float16* wq16 = (_Float16*)(ws + 4194304);              // (768, 256)     384 KB
  _Float16* wp16 = (_Float16*)(ws + 4587520);              // (256, 256)     128 KB
  _Float16* qws  = (_Float16*)(ws + 4718592);              // (b, h, n, d)   4 MB (pre-scaled)
  _Float16* kws  = (_Float16*)(ws + 8912896);              // (b, h, n, d)   4 MB
  _Float16* vws  = (_Float16*)(ws + 13107200);             // (b, c, n)      4 MB
  _Float16* ows  = (_Float16*)(ws + 17301504);             // (b, n, C)      4 MB

  cvt_weights_kernel<<<dim3(768), dim3(256), 0, stream>>>(wqkv, wproj, wq16, wp16);
  gn_kernel<<<dim3(BATCH * 32), dim3(256), 0, stream>>>(x, gs, gb, hn);
  qkv_gemm_kernel<<<dim3(6, 32, BATCH), dim3(256), 0, stream>>>(hn, wq16, qws, kws, vws);
  flash_attn_kernel<<<dim3(32, HEADS, BATCH), dim3(256), 0, stream>>>(qws, kws, vws, ows);
  proj_gemm_kernel<<<dim3(2, 32, BATCH), dim3(256), 0, stream>>>(ows, wp16, bproj, x, out);
}
